// StandardTransformerModel_833223655612
// MI455X (gfx1250) — matrix operand order repaired and checked by reading, not compiled, not run on hardware
//
#include <hip/hip_runtime.h>
#include <hip/hip_bf16.h>

// ---------------------------------------------------------------------------
// MI455X (gfx1250) sliding-window transformer, bf16 WMMA + f32 accumulate.
// ~430 GFLOP total vs ~a few MB of weight traffic -> matrix-core bound.
// One workgroup (8 wave32) per window; all per-window state in LDS (~62 KB).
// Window tile is DMA'd into LDS by the Tensor Data Mover (tensor_load_to_lds,
// TENSORcnt) with LDS row padding done by the TDM pad fields.
// ---------------------------------------------------------------------------

typedef __bf16 bf16_t;
typedef __attribute__((ext_vector_type(8)))  bf16_t   v8bf;
typedef __attribute__((ext_vector_type(16))) bf16_t   v16bf;
typedef __attribute__((ext_vector_type(8)))  float    v8f;
typedef __attribute__((ext_vector_type(4)))  unsigned v4u;
typedef __attribute__((ext_vector_type(8)))  int      v8i_t;
typedef __attribute__((ext_vector_type(4)))  int      v4i_t;

#define EMBD 256
#define FFD  1024
#define WIN  64
#define NHEAD 8
#define HDIM 32
#define NLAYER 2
#define SEQ 1024

// LDS leading dims (elements); all row strides are multiples of 8 elems (16B)
#define XS_LD 264
#define QH_LD 40
#define VT_LD 72
#define SC_LD 72
#define OH_LD 40
#define HC_LD 72

__device__ __forceinline__ v16bf join16(v8bf lo, v8bf hi) {
  return __builtin_shufflevector(lo, hi, 0,1,2,3,4,5,6,7,8,9,10,11,12,13,14,15);
}

// A-fragment (16x32 bf16, M x K): lane L<16 holds row M=L, K=0..7 & 16..23;
// lane L+16 holds row M=L, K=8..15 & 24..31  (ISA 7.12.2)
__device__ __forceinline__ v16bf load_a(const bf16_t* base, int ld, int m0, int k0, int lane) {
  const int row = m0 + (lane & 15);
  const int kb  = k0 + (lane >> 4) * 8;
  const bf16_t* p = base + row * ld + kb;
  return join16(*(const v8bf*)p, *(const v8bf*)(p + 16));
}

// B-fragment (32x16 bf16, K x N) sourced from W[n][k] row-major:
// lane L holds column N = L&15, K = 8*(L>>4) + 0..7 & 16 + 8*(L>>4) + 0..7 (matches A)
__device__ __forceinline__ v16bf load_b(const bf16_t* base, int ld, int n0, int k0, int lane) {
  const int n  = n0 + (lane & 15);
  const int kb = k0 + (lane >> 4) * 8;
  const bf16_t* p = base + n * ld + kb;
  return join16(*(const v8bf*)p, *(const v8bf*)(p + 16));
}

__device__ __forceinline__ v8f wmma_bf16(v16bf a, v16bf b, v8f c) {
  return __builtin_amdgcn_wmma_f32_16x16x32_bf16(false, a, false, b, (short)0, c, false, false);
}

// ---------------------------------------------------------------------------
__global__ void cvt_kernel(const float* __restrict__ src, bf16_t* __restrict__ dst, int n) {
  int i = blockIdx.x * blockDim.x + threadIdx.x;
  const int stride = gridDim.x * blockDim.x;
  for (; i < n; i += stride) dst[i] = (bf16_t)src[i];
}

// emb[2048,256] = inputs[2048,64] @ embed_w[256,64]^T + embed_b   (WMMA)
__global__ __launch_bounds__(256)
void embed_kernel(const float* __restrict__ x, const float* __restrict__ w,
                  const float* __restrict__ bias, bf16_t* __restrict__ emb) {
  const int tid = threadIdx.x, lane = tid & 31, wv = tid >> 5;
  const int l16 = lane & 15, lh = lane >> 4;
  const int t  = blockIdx.x * 8 + wv;        // 0..2047 tiles
  const int mT = (t & 127) * 16;             // over B*S
  const int nT = (t >> 7) * 16;              // over E
  v8f acc = {};
#pragma unroll
  for (int kt = 0; kt < 2; ++kt) {
    const float* pa = x + (mT + l16) * 64 + kt * 32 + lh * 8;
    v16bf a;
#pragma unroll
    for (int u = 0; u < 8; ++u) a[u] = (bf16_t)pa[u];
#pragma unroll
    for (int u = 0; u < 8; ++u) a[8 + u] = (bf16_t)pa[16 + u];
    const float* pb = w + (nT + l16) * 64 + kt * 32 + lh * 8;
    v16bf b;
#pragma unroll
    for (int u = 0; u < 16; ++u) b[u] = (bf16_t)pb[(u & 7) + (u >> 3) * 16];
    acc = wmma_bf16(a, b, acc);
  }
  const float bv = bias[nT + l16];
#pragma unroll
  for (int r = 0; r < 8; ++r)
    emb[(mT + lh * 8 + r) * EMBD + nT + l16] = (bf16_t)(acc[r] + bv);
}

// ---------------------------------------------------------------------------
__global__ __launch_bounds__(256)
void txf_window_kernel(const bf16_t* __restrict__ emb,
                       const bf16_t* __restrict__ wqkv,   // [2][768][256]
                       const bf16_t* __restrict__ wattn,  // [2][256][256]
                       const bf16_t* __restrict__ wff1,   // [2][1024][256]
                       const bf16_t* __restrict__ wff2,   // [2][256][1024]
                       const float* __restrict__ qkv_b,
                       const float* __restrict__ attn_out_b,
                       const float* __restrict__ ln1_g, const float* __restrict__ ln1_b,
                       const float* __restrict__ ffn1_b, const float* __restrict__ ffn2_b,
                       const float* __restrict__ ln2_g, const float* __restrict__ ln2_b,
                       const float* __restrict__ head_w, const float* __restrict__ head_b,
                       float* __restrict__ out) {
  __shared__ __align__(16) char smem[62976];
  bf16_t* xs = (bf16_t*)(smem);             // [64][264] window tile (residual stream)
  bf16_t* qh = (bf16_t*)(smem + 33792);     // [64][40]  q per head
  bf16_t* kh = (bf16_t*)(smem + 38912);     // [64][40]  k per head
  bf16_t* vt = (bf16_t*)(smem + 44032);     // [32][72]  v transposed [hd][key]
  bf16_t* sc = (bf16_t*)(smem + 48640);     // [64][72]  scores/probs (bf16)
  bf16_t* hc = sc;                          // FFN chunk stage aliases sc
  bf16_t* oh = (bf16_t*)(smem + 57856);     // [64][40]  per-head output

  const int tid  = threadIdx.x;
  const int lane = tid & 31;
  const int wv   = tid >> 5;
  const int l16  = lane & 15;
  const int lh   = lane >> 4;

  const int wi = blockIdx.x;                // window index in [0, B*S)
  const int bb = wi >> 10;
  const int ss = wi & (SEQ - 1);

  // ---- gather window into LDS via the Tensor Data Mover ----
  // Valid token rows [zrows, 64) come in as one padded 2D TDM tile:
  // 512B rows in memory -> 528B rows in LDS via pad_interval=512B, pad=16B.
  {
    const int nvalid = (ss + 1 < WIN) ? (ss + 1) : WIN;   // rows with pos >= 0
    const int zrows  = WIN - nvalid;
    if (zrows > 0) {                                      // left zero-padding
      for (int idx = tid; idx < zrows * XS_LD; idx += 256)
        xs[idx] = (bf16_t)0.0f;
    }
    if (wv == 0) {
      const unsigned lds_base =
          (unsigned)(size_t)(void*)(xs + zrows * XS_LD);
      const unsigned long long ga = (unsigned long long)(size_t)
          (emb + ((bb << 10) + (ss - (nvalid - 1))) * EMBD);
      v4u  g0;
      v8i_t g1;
      v4i_t z4 = {};
      v8i_t z8 = {};
      g0[0] = 1u;                                         // count=1 (valid D#)
      g0[1] = lds_base;                                   // lds_addr (bytes)
      g0[2] = (unsigned)(ga & 0xffffffffu);               // global_addr[31:0]
      g0[3] = (unsigned)((ga >> 32) & 0x01ffffffu)        // global_addr[56:32]
            | (2u << 30);                                 // type = 2 ("image")
      g1[0] = (1 << 16)                                   // data_size = 2 bytes
            | (1 << 20)                                   // pad_enable
            | (6 << 22)                                   // pad_interval: 512 B
            | (3 << 25);                                  // pad_amount: 16 B
      g1[1] = (EMBD & 0xffff) << 16;                      // tensor_dim0 = 256
      g1[2] = (nvalid & 0xffff) << 16;                    // tensor_dim1
      g1[3] = (EMBD & 0xffff) << 16;                      // tile_dim0 = 256
      g1[4] = (nvalid & 0xffff);                          // tile_dim1, tile_dim2=0
      g1[5] = EMBD;                                       // tensor_dim0_stride
      g1[6] = 0;
      g1[7] = 0;
      __builtin_amdgcn_tensor_load_to_lds(g0, g1, z4, z4, z8, 0);
      __builtin_amdgcn_s_wait_tensorcnt(0);
    }
  }
  __syncthreads();

  const float scale = 0.17677669529663688f; // 1/sqrt(hd=32)

  for (int l = 0; l < NLAYER; ++l) {
    const bf16_t* Wqkv = wqkv + l * 3 * EMBD * EMBD;
    const bf16_t* Wo   = wattn + l * EMBD * EMBD;
    const bf16_t* W1   = wff1 + l * FFD * EMBD;
    const bf16_t* W2   = wff2 + l * EMBD * FFD;

    // wave's slice of the [64,256] output accumulators
    const int mT = (wv & 3) * 16;
    const int nB = (wv >> 2) * 128;

    v8f oacc[8];
#pragma unroll
    for (int j = 0; j < 8; ++j) oacc[j] = (v8f){};

    for (int h = 0; h < NHEAD; ++h) {
      // ---- QKV for this head: [64, 96], K = 256 ----
      for (int t = wv; t < 24; t += 8) {
        const int m  = (t & 3) * 16;
        const int nc = t >> 2;              // 0..5: q,q,k,k,v,v
        int wrow;
        if (nc < 2)      wrow = h * HDIM + nc * 16;
        else if (nc < 4) wrow = EMBD + h * HDIM + (nc - 2) * 16;
        else             wrow = 2 * EMBD + h * HDIM + (nc - 4) * 16;
        v8f acc = {};
#pragma unroll
        for (int kt = 0; kt < 8; ++kt) {
          v16bf a = load_a(xs, XS_LD, m, kt * 32, lane);
          v16bf b = load_b(Wqkv, EMBD, wrow, kt * 32, lane);
          acc = wmma_bf16(a, b, acc);
        }
        const float bv = qkv_b[l * 3 * EMBD + wrow + l16];
        if (nc < 4) {
          bf16_t* dst = (nc < 2) ? qh : kh;
          const int nloc = (nc & 1) * 16 + l16;
#pragma unroll
          for (int r = 0; r < 8; ++r)
            dst[(m + lh * 8 + r) * QH_LD + nloc] = (bf16_t)(acc[r] + bv);
        } else {                            // v stored transposed [hd][key]
          const int nloc = (nc - 4) * 16 + l16;
#pragma unroll
          for (int r = 0; r < 8; ++r)
            vt[nloc * VT_LD + m + lh * 8 + r] = (bf16_t)(acc[r] + bv);
        }
      }
      __syncthreads();

      // ---- scores = (q k^T) * scale : [64,64], K = 32 ----
      for (int t = wv * 2; t < wv * 2 + 2; ++t) {
        const int m = (t & 3) * 16, n = (t >> 2) * 16;
        v16bf a = load_a(qh, QH_LD, m, 0, lane);
        v16bf b = load_b(kh, QH_LD, n, 0, lane);
        v8f acc = {};
        acc = wmma_bf16(a, b, acc);
#pragma unroll
        for (int r = 0; r < 8; ++r)
          sc[(m + lh * 8 + r) * SC_LD + n + l16] = (bf16_t)(acc[r] * scale);
      }
      __syncthreads();

      // ---- softmax over keys (row per thread) ----
      if (tid < 64) {
        bf16_t* row = sc + tid * SC_LD;
        float mx = -3.0e38f;
        for (int k2 = 0; k2 < WIN; ++k2) { float v = (float)row[k2]; mx = v > mx ? v : mx; }
        float sum = 0.f;
        for (int k2 = 0; k2 < WIN; ++k2) {
          float e = __expf((float)row[k2] - mx);
          sum += e;
          row[k2] = (bf16_t)e;
        }
        const float inv = 1.0f / sum;
        for (int k2 = 0; k2 < WIN; ++k2)
          row[k2] = (bf16_t)((float)row[k2] * inv);
      }
      __syncthreads();

      // ---- o_h = P @ V : [64,32], K = 64 (B sourced from vt[hd][key]) ----
      {
        const int m = (wv & 3) * 16, n = (wv >> 2) * 16;
        v8f acc = {};
#pragma unroll
        for (int kt = 0; kt < 2; ++kt) {
          v16bf a = load_a(sc, SC_LD, m, kt * 32, lane);
          v16bf b = load_b(vt, VT_LD, n, kt * 32, lane);
          acc = wmma_bf16(a, b, acc);
        }
#pragma unroll
        for (int r = 0; r < 8; ++r)
          oh[(m + lh * 8 + r) * OH_LD + n + l16] = (bf16_t)acc[r];
      }
      __syncthreads();

      // ---- fold head into output projection: out += o_h @ Wo[:, h*32:+32]^T ----
      {
        v16bf a = load_a(oh, OH_LD, mT, 0, lane);
#pragma unroll
        for (int j = 0; j < 8; ++j) {
          v16bf b = load_b(Wo, EMBD, nB + j * 16, h * HDIM, lane);
          oacc[j] = wmma_bf16(a, b, oacc[j]);
        }
      }
      __syncthreads();
    } // heads

    // ---- bias + residual (each lane owns its fragment elements) ----
#pragma unroll
    for (int j = 0; j < 8; ++j) {
      const int n  = nB + j * 16 + l16;
      const float bv = attn_out_b[l * EMBD + n];
#pragma unroll
      for (int r = 0; r < 8; ++r) {
        const int m = mT + lh * 8 + r;
        const float v = oacc[j][r] + bv + (float)xs[m * XS_LD + n];
        xs[m * XS_LD + n] = (bf16_t)v;
      }
    }
    __syncthreads();

    // ---- LayerNorm 1 (row per thread, in place) ----
    if (tid < 64) {
      bf16_t* row = xs + tid * XS_LD;
      float mu = 0.f;
      for (int e = 0; e < EMBD; ++e) mu += (float)row[e];
      mu *= (1.0f / EMBD);
      float var = 0.f;
      for (int e = 0; e < EMBD; ++e) { float d = (float)row[e] - mu; var += d * d; }
      var *= (1.0f / EMBD);
      const float inv = __frsqrt_rn(var + 1e-5f);
      for (int e = 0; e < EMBD; ++e) {
        const float v = ((float)row[e] - mu) * inv * ln1_g[l * EMBD + e] + ln1_b[l * EMBD + e];
        row[e] = (bf16_t)v;
      }
    }
    __syncthreads();

    // ---- FFN, chunked over FF (16 chunks of 64), second GEMM folded ----
    v8f facc[8];
#pragma unroll
    for (int j = 0; j < 8; ++j) facc[j] = (v8f){};

    for (int c = 0; c < 16; ++c) {
      if (c + 1 < 16) __builtin_prefetch((const void*)(W1 + (c + 1) * 64 * EMBD), 0, 1);
      // h1_c = silu(x @ W1_c^T + b1_c) : [64,64], K = 256
      for (int t = wv * 2; t < wv * 2 + 2; ++t) {
        const int m = (t & 3) * 16, n = (t >> 2) * 16;
        v8f acc = {};
#pragma unroll
        for (int kt = 0; kt < 8; ++kt) {
          v16bf a = load_a(xs, XS_LD, m, kt * 32, lane);
          v16bf b = load_b(W1, EMBD, c * 64 + n, kt * 32, lane);
          acc = wmma_bf16(a, b, acc);
        }
        const float bv = ffn1_b[l * FFD + c * 64 + n + l16];
#pragma unroll
        for (int r = 0; r < 8; ++r) {
          float v = acc[r] + bv;
          v = v / (1.0f + __expf(-v));      // SiLU
          hc[(m + lh * 8 + r) * HC_LD + n + l16] = (bf16_t)v;
        }
      }
      __syncthreads();

      // out2 += h1_c @ W2[:, c*64:+64]^T  (K = 64)
      {
        v16bf a0 = load_a(hc, HC_LD, mT, 0, lane);
        v16bf a1 = load_a(hc, HC_LD, mT, 32, lane);
#pragma unroll
        for (int j = 0; j < 8; ++j) {
          v16bf b0 = load_b(W2, FFD, nB + j * 16, c * 64, lane);
          facc[j] = wmma_bf16(a0, b0, facc[j]);
          v16bf b1 = load_b(W2, FFD, nB + j * 16, c * 64 + 32, lane);
          facc[j] = wmma_bf16(a1, b1, facc[j]);
        }
      }
      __syncthreads();
    }

    // ---- bias + residual + LayerNorm 2 ----
#pragma unroll
    for (int j = 0; j < 8; ++j) {
      const int n  = nB + j * 16 + l16;
      const float bv = ffn2_b[l * EMBD + n];
#pragma unroll
      for (int r = 0; r < 8; ++r) {
        const int m = mT + lh * 8 + r;
        const float v = facc[j][r] + bv + (float)xs[m * XS_LD + n];
        xs[m * XS_LD + n] = (bf16_t)v;
      }
    }
    __syncthreads();

    if (tid < 64) {
      bf16_t* row = xs + tid * XS_LD;
      float mu = 0.f;
      for (int e = 0; e < EMBD; ++e) mu += (float)row[e];
      mu *= (1.0f / EMBD);
      float var = 0.f;
      for (int e = 0; e < EMBD; ++e) { float d = (float)row[e] - mu; var += d * d; }
      var *= (1.0f / EMBD);
      const float inv = __frsqrt_rn(var + 1e-5f);
      for (int e = 0; e < EMBD; ++e) {
        const float v = ((float)row[e] - mu) * inv * ln2_g[l * EMBD + e] + ln2_b[l * EMBD + e];
        row[e] = (bf16_t)v;
      }
    }
    __syncthreads();
  } // layers

  // ---- classifier head on last window row ----
  if (tid < 10) {
    float o = head_b[tid];
    const bf16_t* row = xs + 63 * XS_LD;
    for (int e = 0; e < EMBD; ++e) o += (float)row[e] * head_w[tid * EMBD + e];
    out[wi * 10 + tid] = o;
  }
}

// ---------------------------------------------------------------------------
extern "C" void kernel_launch(void* const* d_in, const int* in_sizes, int n_in,
                              void* d_out, int out_size, void* d_ws, size_t ws_size,
                              hipStream_t stream) {
  const float* inputs  = (const float*)d_in[0];
  const float* embed_w = (const float*)d_in[1];
  const float* embed_b = (const float*)d_in[2];
  const float* qkv_w   = (const float*)d_in[3];
  const float* qkv_b   = (const float*)d_in[4];
  const float* attn_w  = (const float*)d_in[5];
  const float* attn_b  = (const float*)d_in[6];
  const float* ln1_g   = (const float*)d_in[7];
  const float* ln1_b   = (const float*)d_in[8];
  const float* ffn1_w  = (const float*)d_in[9];
  const float* ffn1_b  = (const float*)d_in[10];
  const float* ffn2_w  = (const float*)d_in[11];
  const float* ffn2_b  = (const float*)d_in[12];
  const float* ln2_g   = (const float*)d_in[13];
  const float* ln2_b   = (const float*)d_in[14];
  const float* head_w  = (const float*)d_in[15];
  const float* head_b  = (const float*)d_in[16];

  bf16_t* ws    = (bf16_t*)d_ws;
  bf16_t* emb   = ws;                 // 2048*256      = 524288 elems
  bf16_t* wqkv  = ws + 524288;        // 2*768*256     = 393216
  bf16_t* wattn = ws + 917504;        // 2*256*256     = 131072
  bf16_t* wff1  = ws + 1048576;       // 2*1024*256    = 524288
  bf16_t* wff2  = ws + 1572864;       // 2*256*1024    = 524288
                                      // total 4 MB of workspace

  cvt_kernel<<<384, 256, 0, stream>>>(qkv_w,  wqkv,  393216);
  cvt_kernel<<<128, 256, 0, stream>>>(attn_w, wattn, 131072);
  cvt_kernel<<<512, 256, 0, stream>>>(ffn1_w, wff1,  524288);
  cvt_kernel<<<512, 256, 0, stream>>>(ffn2_w, wff2,  524288);

  embed_kernel<<<256, 256, 0, stream>>>(inputs, embed_w, embed_b, emb);

  txf_window_kernel<<<2048, 256, 0, stream>>>(
      emb, wqkv, wattn, wff1, wff2,
      qkv_b, attn_b, ln1_g, ln1_b, ffn1_b, ffn2_b, ln2_g, ln2_b,
      head_w, head_b, (float*)d_out);
}
